// TransformerSegmentation_30116310679606
// MI455X (gfx1250) — compile-verified, matrix operand order checked by reading, not run on hardware
//
#include <hip/hip_runtime.h>
#include <hip/hip_bf16.h>
#include <math.h>

// ---------------------------------------------------------------------------
// MI455X / gfx1250 ViT forward. wave32, WMMA bf16 16x16x32, fp32 accumulate.
// GEMMs register-blocked 64x64 per wave (4x4 WMMA tiles, 16 wmma / 8 loads).
// ---------------------------------------------------------------------------

typedef __attribute__((ext_vector_type(16))) __bf16 v16bf;
typedef __attribute__((ext_vector_type(8)))  float  v8f;

#define DEPTH 12
#define DMODEL 768
#define FMLP 3072
#define NHEAD 12
#define HDIM 64
#define NTOK 1024           // 32x32 patches
#define BATCH 2
#define MROWS (BATCH * NTOK) // 2048
#define KPATCH 1536          // 6*16*16

static __device__ __forceinline__ float gelu_exact(float x) {
    return 0.5f * x * (1.0f + erff(x * 0.70710678118654752f));
}

// -------------------------------- conversions ------------------------------
__global__ void cvt_bf16_kernel(const float* __restrict__ src,
                                __bf16* __restrict__ dst, size_t n) {
    size_t i = (size_t)blockIdx.x * blockDim.x + threadIdx.x;
    if (i < n) dst[i] = (__bf16)src[i];
}

// im2col of x:(2,6,512,512) -> A[2048][1536] bf16, k = c*256 + p*16 + q
__global__ void im2col_kernel(const float* __restrict__ x,
                              __bf16* __restrict__ A) {
    size_t i = (size_t)blockIdx.x * blockDim.x + threadIdx.x;
    const size_t total = (size_t)MROWS * KPATCH;
    if (i >= total) return;
    int k   = (int)(i % KPATCH);
    int tok = (int)(i / KPATCH);
    int b = tok >> 10, n = tok & 1023;
    int hp = n >> 5, wp = n & 31;
    int c = k >> 8, p = (k >> 4) & 15, q = k & 15;
    size_t xi = (((size_t)(b * 6 + c) * 512) + hp * 16 + p) * 512 + wp * 16 + q;
    A[i] = (__bf16)x[xi];
}

// ------------------------------ WMMA GEMM ----------------------------------
// C[m,n] = sum_k A[m,k] * W[n,k]  (+bias[n]) (+resid) (gelu?) -> outf / outbf
// Each wave computes a 64x64 supertile = 4x4 WMMA tiles. Per K-step of 32:
// 4 A-frags + 4 B-frags (8 x 32B loads) feed 16 v_wmma ops.
// A, W row-major, K contiguous, K%32==0, M%64==0, N%64==0.
__global__ __launch_bounds__(128)
void wmma_gemm_bf16(const __bf16* __restrict__ A, const __bf16* __restrict__ W,
                    const float* __restrict__ bias, const float* __restrict__ resid,
                    float* __restrict__ outf, __bf16* __restrict__ outbf,
                    int M, int N, int K, int act) {
    int lane = threadIdx.x & 31;
    int wave = threadIdx.x >> 5;
    int tile = blockIdx.x * (blockDim.x >> 5) + wave;
    int ntn = N >> 6;
    int tm = tile / ntn, tn = tile % ntn;
    if (tm >= (M >> 6)) return;
    int l16 = lane & 15, half = lane >> 4;

    const __bf16* arow = A + (size_t)(tm * 64 + l16) * K + half * 16;
    const __bf16* wrow = W + (size_t)(tn * 64 + l16) * K + half * 16;
    const size_t rstep = (size_t)16 * K;   // 16 rows ahead

    v8f acc[4][4] = {};
    for (int k0 = 0; k0 < K; k0 += 32) {
        v16bf a[4], b[4];
#pragma unroll
        for (int i = 0; i < 4; ++i) {
            __builtin_prefetch(arow + i * rstep + k0 + 256, 0, 1); // global_prefetch_b8
            a[i] = *(const v16bf*)(arow + i * rstep + k0);
        }
#pragma unroll
        for (int j = 0; j < 4; ++j) {
            __builtin_prefetch(wrow + j * rstep + k0 + 256, 0, 1);
            b[j] = *(const v16bf*)(wrow + j * rstep + k0);
        }
#pragma unroll
        for (int i = 0; i < 4; ++i)
#pragma unroll
            for (int j = 0; j < 4; ++j)
                acc[i][j] = __builtin_amdgcn_wmma_f32_16x16x32_bf16(
                    false, a[i], false, b[j], (short)0, acc[i][j], false, false);
    }

    float bb[4];
#pragma unroll
    for (int j = 0; j < 4; ++j)
        bb[j] = bias ? bias[tn * 64 + j * 16 + l16] : 0.0f;

#pragma unroll
    for (int i = 0; i < 4; ++i) {
#pragma unroll
        for (int r = 0; r < 8; ++r) {
            int mrow = tm * 64 + i * 16 + r + 8 * half; // C-frag: VGPR r -> M=r+8*half
            size_t base = (size_t)mrow * N + tn * 64 + l16;
#pragma unroll
            for (int j = 0; j < 4; ++j) {
                size_t idx = base + j * 16;
                float v = acc[i][j][r] + bb[j];
                if (resid) v += resid[idx];
                if (act)   v = gelu_exact(v);
                if (outf)  outf[idx]  = v;
                if (outbf) outbf[idx] = (__bf16)v;
            }
        }
    }
}

// ------------------------------ LayerNorm ----------------------------------
// One wave (32 threads) per row of D=768. Optional +pos_embed, f32/bf16 outs.
__global__ void ln_kernel(const float* __restrict__ in,
                          const float* __restrict__ g, const float* __restrict__ b,
                          const float* __restrict__ pos,
                          float* __restrict__ outf, __bf16* __restrict__ outbf,
                          int D) {
    int row = blockIdx.x;
    int lane = threadIdx.x;
    const float* x = in + (size_t)row * D;
    float s = 0.f, s2 = 0.f;
    for (int i = lane; i < D; i += 32) { float v = x[i]; s += v; s2 += v * v; }
#pragma unroll
    for (int off = 1; off < 32; off <<= 1) {
        s  += __shfl_xor(s,  off, 32);
        s2 += __shfl_xor(s2, off, 32);
    }
    float mean = s / D;
    float var  = s2 / D - mean * mean;
    float rstd = rsqrtf(var + 1e-5f);
    int pn = row & (NTOK - 1);
    for (int i = lane; i < D; i += 32) {
        float v = (x[i] - mean) * rstd * g[i] + b[i];
        if (pos)   v += pos[(size_t)pn * D + i];
        size_t idx = (size_t)row * D + i;
        if (outf)  outf[idx]  = v;
        if (outbf) outbf[idx] = (__bf16)v;
    }
}

// --------------------------- fused flash attention -------------------------
// qkv_bf: [B*N][2304] bf16 (q|k|v, each NHEAD*HDIM). One wave per (b,h,16-row
// q block). Online softmax; scores & P*V all via WMMA bf16.
static __device__ __forceinline__ float rmax16(float v) {
#pragma unroll
    for (int off = 1; off < 16; off <<= 1) v = fmaxf(v, __shfl_xor(v, off, 32));
    return v;
}
static __device__ __forceinline__ float rsum16(float v) {
#pragma unroll
    for (int off = 1; off < 16; off <<= 1) v += __shfl_xor(v, off, 32);
    return v;
}

__global__ __launch_bounds__(128)
void attn_kernel(const __bf16* __restrict__ qkv_bf, __bf16* __restrict__ o_bf) {
    __shared__ __bf16 ldsP[4][16 * 32];  // per-wave P staging (C-frag -> A-frag)
    int lane = threadIdx.x & 31;
    int wave = threadIdx.x >> 5;
    int gid  = blockIdx.x * 4 + wave;          // b * NHEAD*64 + h*64 + qt
    int b  = gid / (NHEAD * 64);
    int r0 = gid % (NHEAD * 64);
    int h  = r0 / 64;
    int qt = r0 % 64;
    int l16 = lane & 15, half = lane >> 4;
    const float scale = 0.125f;                // 1/sqrt(64)

    const __bf16* qkv = qkv_bf + (size_t)b * NTOK * 2304;

    // Q A-fragments (16x64 split as two 16x32)
    const __bf16* qrow = qkv + (size_t)(qt * 16 + l16) * 2304 + h * HDIM;
    v16bf aq0 = *(const v16bf*)(qrow + half * 16);
    v16bf aq1 = *(const v16bf*)(qrow + 32 + half * 16);

    float m[8], l[8];
    v8f accO[4];
#pragma unroll
    for (int r = 0; r < 8; ++r) { m[r] = -3.0e38f; l[r] = 0.f; }
#pragma unroll
    for (int t = 0; t < 4; ++t) { v8f z = {}; accO[t] = z; }

    for (int jt = 0; jt < NTOK; jt += 32) {
        // ---- scores S = Q*K^T for 32 keys (two 16x16 tiles) ----
        v8f s0 = {}, s1 = {};
        {
            const __bf16* krow = qkv + (size_t)(jt + l16) * 2304 + DMODEL + h * HDIM;
            v16bf bk0 = *(const v16bf*)(krow + half * 16);
            v16bf bk1 = *(const v16bf*)(krow + 32 + half * 16);
            s0 = __builtin_amdgcn_wmma_f32_16x16x32_bf16(false, aq0, false, bk0, (short)0, s0, false, false);
            s0 = __builtin_amdgcn_wmma_f32_16x16x32_bf16(false, aq1, false, bk1, (short)0, s0, false, false);
        }
        {
            const __bf16* krow = qkv + (size_t)(jt + 16 + l16) * 2304 + DMODEL + h * HDIM;
            v16bf bk0 = *(const v16bf*)(krow + half * 16);
            v16bf bk1 = *(const v16bf*)(krow + 32 + half * 16);
            s1 = __builtin_amdgcn_wmma_f32_16x16x32_bf16(false, aq0, false, bk0, (short)0, s1, false, false);
            s1 = __builtin_amdgcn_wmma_f32_16x16x32_bf16(false, aq1, false, bk1, (short)0, s1, false, false);
        }
        // ---- online softmax over the 32 new columns ----
#pragma unroll
        for (int r = 0; r < 8; ++r) {
            float a0 = s0[r] * scale;
            float a1 = s1[r] * scale;
            float rm = rmax16(fmaxf(a0, a1));
            float mn = fmaxf(m[r], rm);
            float alpha = __expf(m[r] - mn);
            float p0 = __expf(a0 - mn);
            float p1 = __expf(a1 - mn);
            l[r] = l[r] * alpha + rsum16(p0 + p1);
            m[r] = mn;
#pragma unroll
            for (int t = 0; t < 4; ++t) accO[t][r] *= alpha;
            int prow = r + 8 * half;
            ldsP[wave][prow * 32 + l16]      = (__bf16)p0;
            ldsP[wave][prow * 32 + 16 + l16] = (__bf16)p1;
        }
        // reload P as an A-fragment (same-wave DS ops are in order)
        v16bf ap = *(const v16bf*)(&ldsP[wave][l16 * 32 + half * 16]);

        // ---- O += P(16x32) * V(32x64) : 4 WMMAs over dim chunks ----
#pragma unroll
        for (int t = 0; t < 4; ++t) {
            const __bf16* vbase = qkv + 2 * DMODEL + h * HDIM + t * 16 + l16;
            v16bf bv;
#pragma unroll
            for (int i = 0; i < 16; ++i)
                bv[i] = vbase[(size_t)(jt + half * 16 + i) * 2304];
            accO[t] = __builtin_amdgcn_wmma_f32_16x16x32_bf16(
                false, ap, false, bv, (short)0, accO[t], false, false);
        }
    }

    // ---- normalize + write O (bf16, layout [B*N][D]) ----
#pragma unroll
    for (int r = 0; r < 8; ++r) {
        int tokrow = qt * 16 + r + 8 * half;
        float inv = 1.0f / l[r];
        size_t base = ((size_t)b * NTOK + tokrow) * DMODEL + h * HDIM;
#pragma unroll
        for (int t = 0; t < 4; ++t)
            o_bf[base + t * 16 + l16] = (__bf16)(accO[t][r] * inv);
    }
}

// ------------------------------ tiny head ----------------------------------
__global__ void head2_kernel(const float* __restrict__ h1,
                             const float* __restrict__ w2,
                             const float* __restrict__ b2,
                             float* __restrict__ seg) {
    int idx = blockIdx.x * blockDim.x + threadIdx.x;   // 2048*2
    if (idx >= MROWS * 2) return;
    int tok = idx >> 1, c = idx & 1;
    const float* hr = h1 + (size_t)tok * 384;
    const float* wr = w2 + (size_t)c * 384;
    float s = b2[c];
    for (int e = 0; e < 384; ++e) s += hr[e] * wr[e];
    seg[idx] = s;
}

// seg[2048][2] ((b, y*32+x, c)) -> out (B,2,512,512), half-pixel bilinear
__global__ void resize_kernel(const float* __restrict__ seg,
                              float* __restrict__ out) {
    int idx = blockIdx.x * blockDim.x + threadIdx.x;
    const int total = BATCH * 2 * 512 * 512;
    if (idx >= total) return;
    int xo = idx & 511;
    int yo = (idx >> 9) & 511;
    int c  = (idx >> 18) & 1;
    int b  = idx >> 19;
    float sx = (xo + 0.5f) * (32.0f / 512.0f) - 0.5f;
    float sy = (yo + 0.5f) * (32.0f / 512.0f) - 0.5f;
    float fxf = floorf(sx), fyf = floorf(sy);
    float fx = sx - fxf, fy = sy - fyf;
    int x0 = (int)fxf, y0 = (int)fyf;
    int x1 = x0 + 1, y1 = y0 + 1;
    x0 = min(max(x0, 0), 31); x1 = min(max(x1, 0), 31);
    y0 = min(max(y0, 0), 31); y1 = min(max(y1, 0), 31);
    auto at = [&](int yy, int xx) {
        return seg[((size_t)(b * NTOK + yy * 32 + xx)) * 2 + c];
    };
    float v = (1.f - fy) * ((1.f - fx) * at(y0, x0) + fx * at(y0, x1)) +
              fy        * ((1.f - fx) * at(y1, x0) + fx * at(y1, x1));
    out[idx] = v;
}

// ------------------------------- host side ---------------------------------
extern "C" void kernel_launch(void* const* d_in, const int* in_sizes, int n_in,
                              void* d_out, int out_size, void* d_ws, size_t ws_size,
                              hipStream_t stream) {
    const float* x      = (const float*)d_in[0];
    const float* conv_w = (const float*)d_in[1];
    const float* conv_b = (const float*)d_in[2];
    const float* pe_g   = (const float*)d_in[3];
    const float* pe_b   = (const float*)d_in[4];
    const float* pos    = (const float*)d_in[5];
    const float* ln1_g  = (const float*)d_in[6];
    const float* ln1_b  = (const float*)d_in[7];
    const float* qkv_w  = (const float*)d_in[8];
    const float* qkv_b  = (const float*)d_in[9];
    const float* proj_w = (const float*)d_in[10];
    const float* proj_b = (const float*)d_in[11];
    const float* ln2_g  = (const float*)d_in[12];
    const float* ln2_b  = (const float*)d_in[13];
    const float* fc1_w  = (const float*)d_in[14];
    const float* fc1_b  = (const float*)d_in[15];
    const float* fc2_w  = (const float*)d_in[16];
    const float* fc2_b  = (const float*)d_in[17];
    const float* fn_g   = (const float*)d_in[18];
    const float* fn_b   = (const float*)d_in[19];
    const float* h1_w   = (const float*)d_in[20];
    const float* h1_b   = (const float*)d_in[21];
    const float* h2_w   = (const float*)d_in[22];
    const float* h2_b   = (const float*)d_in[23];
    float* out = (float*)d_out;

    // ---- workspace bump allocator (needs ~230 MB) ----
    char* ws = (char*)d_ws;
    size_t off = 0;
    auto take = [&](size_t bytes) -> char* {
        char* p = ws + off;
        off = (off + bytes + 255) & ~(size_t)255;
        return p;
    };
    const size_t SZ_QKVW = (size_t)DEPTH * 3 * DMODEL * DMODEL; // 21.2M
    const size_t SZ_PRJW = (size_t)DEPTH * DMODEL * DMODEL;     // 7.08M
    const size_t SZ_FCW  = (size_t)DEPTH * FMLP * DMODEL;       // 28.3M
    const size_t SZ_CONV = (size_t)DMODEL * KPATCH;             // 1.18M
    const size_t SZ_H1W  = (size_t)384 * DMODEL;

    __bf16* wq  = (__bf16*)take(SZ_QKVW * 2);
    __bf16* wp  = (__bf16*)take(SZ_PRJW * 2);
    __bf16* w1  = (__bf16*)take(SZ_FCW * 2);
    __bf16* w2  = (__bf16*)take(SZ_FCW * 2);
    __bf16* wc  = (__bf16*)take(SZ_CONV * 2);
    __bf16* wh1 = (__bf16*)take(SZ_H1W * 2);

    __bf16* Aim    = (__bf16*)take((size_t)MROWS * KPATCH * 2);
    float*  tokraw = (float*) take((size_t)MROWS * DMODEL * 4);
    float*  tok    = (float*) take((size_t)MROWS * DMODEL * 4);
    __bf16* hbf    = (__bf16*)take((size_t)MROWS * DMODEL * 2);
    __bf16* qkvbf  = (__bf16*)take((size_t)MROWS * 3 * DMODEL * 2);
    __bf16* obf    = (__bf16*)take((size_t)MROWS * DMODEL * 2);
    __bf16* midbf  = (__bf16*)take((size_t)MROWS * FMLP * 2);
    float*  h1f    = (float*) take((size_t)MROWS * 384 * 4);
    float*  segf   = (float*) take((size_t)MROWS * 2 * 4);

    auto cvt = [&](const float* s, __bf16* d, size_t n) {
        cvt_bf16_kernel<<<(unsigned)((n + 255) / 256), 256, 0, stream>>>(s, d, n);
    };
    auto gemm = [&](const __bf16* A, const __bf16* W, const float* bias,
                    const float* resid, float* of, __bf16* ob,
                    int M, int N, int K, int act) {
        int tiles = (M / 64) * (N / 64);   // 64x64 supertiles, one per wave
        wmma_gemm_bf16<<<(tiles + 3) / 4, 128, 0, stream>>>(A, W, bias, resid,
                                                            of, ob, M, N, K, act);
    };
    auto ln = [&](const float* in, const float* g, const float* b,
                  const float* p, float* of, __bf16* ob) {
        ln_kernel<<<MROWS, 32, 0, stream>>>(in, g, b, p, of, ob, DMODEL);
    };

    // ---- weight conversion (deterministic, idempotent) ----
    cvt(qkv_w,  wq,  SZ_QKVW);
    cvt(proj_w, wp,  SZ_PRJW);
    cvt(fc1_w,  w1,  SZ_FCW);
    cvt(fc2_w,  w2,  SZ_FCW);
    cvt(conv_w, wc,  SZ_CONV);
    cvt(h1_w,   wh1, SZ_H1W);

    // ---- patch embed: im2col + GEMM + LN(+pos) ----
    {
        size_t n = (size_t)MROWS * KPATCH;
        im2col_kernel<<<(unsigned)((n + 255) / 256), 256, 0, stream>>>(x, Aim);
    }
    gemm(Aim, wc, conv_b, nullptr, tokraw, nullptr, MROWS, DMODEL, KPATCH, 0);
    ln(tokraw, pe_g, pe_b, pos, tok, nullptr);

    // ---- transformer layers ----
    for (int i = 0; i < DEPTH; ++i) {
        ln(tok, ln1_g + i * DMODEL, ln1_b + i * DMODEL, nullptr, nullptr, hbf);
        gemm(hbf, wq + (size_t)i * 3 * DMODEL * DMODEL, qkv_b + i * 3 * DMODEL,
             nullptr, nullptr, qkvbf, MROWS, 3 * DMODEL, DMODEL, 0);
        attn_kernel<<<(BATCH * NHEAD * 64) / 4, 128, 0, stream>>>(qkvbf, obf);
        gemm(obf, wp + (size_t)i * DMODEL * DMODEL, proj_b + i * DMODEL,
             tok, tok, nullptr, MROWS, DMODEL, DMODEL, 0);
        ln(tok, ln2_g + i * DMODEL, ln2_b + i * DMODEL, nullptr, nullptr, hbf);
        gemm(hbf, w1 + (size_t)i * FMLP * DMODEL, fc1_b + i * FMLP,
             nullptr, nullptr, midbf, MROWS, FMLP, DMODEL, 1);
        gemm(midbf, w2 + (size_t)i * DMODEL * FMLP, fc2_b + i * DMODEL,
             tok, tok, nullptr, MROWS, DMODEL, FMLP, 0);
    }

    // ---- head ----
    ln(tok, fn_g, fn_b, nullptr, nullptr, hbf);
    gemm(hbf, wh1, h1_b, nullptr, h1f, nullptr, MROWS, 384, DMODEL, 1);
    head2_kernel<<<(MROWS * 2 + 255) / 256, 256, 0, stream>>>(h1f, h2_w, h2_b, segf);
    {
        int total = BATCH * 2 * 512 * 512;
        resize_kernel<<<(total + 255) / 256, 256, 0, stream>>>(segf, out);
    }
    (void)in_sizes; (void)n_in; (void)out_size; (void)ws_size;
}
